// MultiHeadAttention_62345745269155
// MI455X (gfx1250) — compile-verified, matrix operand order checked by reading, not run on hardware
//
#include <hip/hip_runtime.h>
#include <hip/hip_bf16.h>

// ---------------- problem constants ----------------
#define BB 2
#define TT 2048
#define CC 1024
#define HH 16
#define DH 64
#define MTOK (BB*TT)          // 4096 tokens
#define BH (BB*HH)            // 32 head-batches
#define ATTN_OFF ((size_t)MTOK*CC)   // out elements before attn_weights in d_out

typedef __attribute__((ext_vector_type(16))) __bf16 v16bf;
typedef __attribute__((ext_vector_type(8)))  __bf16 v8bf;
typedef __attribute__((ext_vector_type(8)))  float  v8f;
typedef __attribute__((ext_vector_type(4)))  float  f4;

// ---------------- helpers ----------------
static __device__ __forceinline__ __bf16 f2bf(float f) {
  unsigned u = __builtin_bit_cast(unsigned, f);
  u = u + 0x7FFFu + ((u >> 16) & 1u);           // round-to-nearest-even
  unsigned short h = (unsigned short)(u >> 16);
  return __builtin_bit_cast(__bf16, h);
}

// CDNA5 async global->LDS copy (16B per lane), tracked by ASYNCcnt.
static __device__ __forceinline__ void async_b128(unsigned lds_off, const void* gaddr) {
  asm volatile("global_load_async_to_lds_b128 %0, %1, off"
               :: "v"(lds_off), "v"(gaddr) : "memory");
}
#define WAIT_ASYNC(n) asm volatile("s_wait_asynccnt " #n ::: "memory")

// Fragment loader for 16-bit A/B operands of V_WMMA_F32_16X16X32_BF16.
// ISA layout (16-bit A 16x32): lanes 0-15 hold row M=lane with K={0..7,16..23},
// lanes 16-31 hold row M=lane-16 with K={8..15,24..31}.  B uses the mirrored
// layout with "row" = output column, streaming along the contraction dim, so
// both operands load 2x16B contiguous chunks from row-major [rows x ld] data.
static __device__ __forceinline__ v16bf load_frag_b16(const __bf16* __restrict__ base,
                                                      int ld, int row0, int k0) {
  int lane = threadIdx.x & 31;
  int r    = lane & 15;
  int koff = (lane >> 4) << 3;                  // 0 or 8
  const __bf16* p = base + (size_t)(row0 + r) * ld + k0 + koff;
  union { v16bf v; v8bf h[2]; } u;
  u.h[0] = *(const v8bf*)(p);                   // K = koff .. koff+7
  u.h[1] = *(const v8bf*)(p + 16);              // K = koff+16 .. koff+23
  return u.v;
}

// Same fragment pattern but sourced from an LDS tile with ld == 32 elements.
static __device__ __forceinline__ v16bf lds_frag(const __bf16* base, int row0) {
  int lane = threadIdx.x & 31;
  int r    = lane & 15;
  int koff = (lane >> 4) << 3;
  const __bf16* p = base + (row0 + r) * 32 + koff;
  union { v16bf v; v8bf h[2]; } u;
  u.h[0] = *(const v8bf*)(p);
  u.h[1] = *(const v8bf*)(p + 16);
  return u.v;
}

// LDS fp32 tile (ld == 32 floats) -> bf16 fragment.
static __device__ __forceinline__ v16bf lds_frag_f32(const float* base, int row0) {
  int lane = threadIdx.x & 31;
  int r    = lane & 15;
  int koff = (lane >> 4) << 3;
  const float* p = base + (row0 + r) * 32 + koff;
  float t[16];
  *(f4*)&t[0]  = *(const f4*)(p);
  *(f4*)&t[4]  = *(const f4*)(p + 4);
  *(f4*)&t[8]  = *(const f4*)(p + 16);
  *(f4*)&t[12] = *(const f4*)(p + 20);
  v16bf v;
#pragma unroll
  for (int i = 0; i < 16; ++i) v[i] = f2bf(t[i]);
  return v;
}

#define WMMA_BF16(a, b, c) \
  __builtin_amdgcn_wmma_f32_16x16x32_bf16(false, (a), false, (b), (short)0, (c), false, false)

// ---------------- BatchNorm: per-channel stats -> scale/shift ----------------
__global__ __launch_bounds__(256) void bn_stats_kernel(const float* __restrict__ x,
                                                       const float* __restrict__ gamma,
                                                       const float* __restrict__ beta,
                                                       float* __restrict__ scale,
                                                       float* __restrict__ shift) {
  int c = blockIdx.x, tid = threadIdx.x;
  float s = 0.f, s2 = 0.f;
  for (int i = tid; i < MTOK; i += 256) {
    float v = x[(size_t)i * CC + c];
    s += v; s2 += v * v;
  }
  __shared__ float rs[256], rq[256];
  rs[tid] = s; rq[tid] = s2; __syncthreads();
  for (int st = 128; st > 0; st >>= 1) {
    if (tid < st) { rs[tid] += rs[tid + st]; rq[tid] += rq[tid + st]; }
    __syncthreads();
  }
  if (tid == 0) {
    float mu  = rs[0] * (1.0f / MTOK);
    float var = rq[0] * (1.0f / MTOK) - mu * mu;
    float sc  = gamma[c] * rsqrtf(var + 1e-5f);
    scale[c] = sc;
    shift[c] = beta[c] - mu * sc;
  }
}

__global__ __launch_bounds__(256) void bn_apply_kernel(const float* __restrict__ x,
                                                       const float* __restrict__ scale,
                                                       const float* __restrict__ shift,
                                                       __bf16* __restrict__ xn) {
  size_t i = (size_t)blockIdx.x * 256 + threadIdx.x;
  if (i < (size_t)MTOK * CC) {
    int c = (int)(i & (CC - 1));
    xn[i] = f2bf(x[i] * scale[c] + shift[c]);
  }
}

__global__ __launch_bounds__(256) void cvt_bf16_kernel(const float* __restrict__ src,
                                                       __bf16* __restrict__ dst, int n) {
  int i = blockIdx.x * 256 + threadIdx.x;
  if (i < n) dst[i] = f2bf(src[i]);
}

// ---------------- projection GEMM: Y = XN @ W^T + b ----------------
// A: [MTOK x CC] bf16, W: [CC x CC] bf16 (row n of W == column n of B).
// 128x128 block tile; A/B 128x32 k-slices staged in double-buffered LDS via
// CDNA5 async global->LDS copies (ASYNCcnt), WMMA consumes via ds_load_b128.
// mode 0: write bf16 Q[b,h,t,d]   mode 1: bf16 K[b,h,t,d]
// mode 2: write bf16 V^T[b,h,d,t] mode 3: write fp32 out[m,n]
__global__ __launch_bounds__(256) void gemm_proj_kernel(const __bf16* __restrict__ A,
                                                        const __bf16* __restrict__ W,
                                                        const float* __restrict__ bias,
                                                        void* __restrict__ outp, int mode) {
  __shared__ __bf16 lA[2][128 * 32];   // 8 KiB per buffer
  __shared__ __bf16 lB[2][128 * 32];

  int tid  = threadIdx.x;
  int wave = tid >> 5;
  int lane = tid & 31;
  int mb = blockIdx.x * 128, nb = blockIdx.y * 128;
  int wm = (wave & 3) * 32;            // wave's rows within the block tile
  int wn = (wave >> 2) * 64;           // wave's cols within the block tile

  // Stage one 128x32 k-slice of A and of W into LDS buffer `buf`.
  // 512 16B-chunks per tile, 256 threads -> 2 chunks each per tile (4 async/thread,
  // uniform across the wave so ASYNCcnt waits are exact).
  auto prefetch = [&](int buf, int k0) {
#pragma unroll
    for (int i = 0; i < 2; ++i) {
      int c   = tid * 2 + i;
      int row = c >> 2;
      int col = (c & 3) * 8;
      const __bf16* ga = A + (size_t)(mb + row) * CC + k0 + col;
      async_b128((unsigned)(size_t)&lA[buf][row * 32 + col], ga);
      const __bf16* gb = W + (size_t)(nb + row) * CC + k0 + col;
      async_b128((unsigned)(size_t)&lB[buf][row * 32 + col], gb);
    }
  };

  v8f acc[2][4];
#pragma unroll
  for (int i = 0; i < 2; ++i)
#pragma unroll
    for (int j = 0; j < 4; ++j) acc[i][j] = (v8f)(0.f);

  const int KSTEPS = CC / 32;
  prefetch(0, 0);
  for (int kk = 0; kk < KSTEPS; ++kk) {
    int cur = kk & 1;
    if (kk + 1 < KSTEPS) {
      prefetch(cur ^ 1, (kk + 1) * 32);
      WAIT_ASYNC(0x4);                 // current batch (in-order) complete
    } else {
      WAIT_ASYNC(0x0);
    }
    __syncthreads();                   // everyone's slice visible in LDS

    v16bf a0 = lds_frag(lA[cur], wm);
    v16bf a1 = lds_frag(lA[cur], wm + 16);
    v16bf b[4];
#pragma unroll
    for (int j = 0; j < 4; ++j) b[j] = lds_frag(lB[cur], wn + 16 * j);
#pragma unroll
    for (int j = 0; j < 4; ++j) {
      acc[0][j] = WMMA_BF16(a0, b[j], acc[0][j]);
      acc[1][j] = WMMA_BF16(a1, b[j], acc[1][j]);
    }
    __syncthreads();                   // done reading before buffer reuse
  }

  int m0 = mb + wm, n0 = nb + wn;
  int nlo = lane & 15, rhi = (lane >> 4) * 8;
#pragma unroll
  for (int i = 0; i < 2; ++i) {
#pragma unroll
    for (int j = 0; j < 4; ++j) {
      int n = n0 + 16 * j + nlo;
      float bv = bias[n];
#pragma unroll
      for (int r = 0; r < 8; ++r) {
        int m = m0 + 16 * i + rhi + r;
        float val = acc[i][j][r] + bv;
        if (mode == 3) {
          ((float*)outp)[(size_t)m * CC + n] = val;
        } else {
          int b_ = m >> 11, t = m & (TT - 1);
          int h  = n >> 6,  d = n & (DH - 1);
          __bf16 bb = f2bf(val);
          if (mode == 2)
            ((__bf16*)outp)[(((size_t)(b_ * HH + h)) * DH + d) * TT + t] = bb;  // V^T
          else
            ((__bf16*)outp)[(((size_t)(b_ * HH + h)) * TT + t) * DH + d] = bb;  // Q / K
        }
      }
    }
  }
}

// ---------------- attention scores: S = (Q @ K^T) * 1/sqrt(Dh) ----------------
__global__ __launch_bounds__(256) void scores_kernel(const __bf16* __restrict__ Q,
                                                     const __bf16* __restrict__ Km,
                                                     float* __restrict__ attn) {
  int bh = blockIdx.z;
  const __bf16* Qb = Q  + (size_t)bh * TT * DH;
  const __bf16* Kb = Km + (size_t)bh * TT * DH;
  float* Sb = attn + (size_t)bh * TT * TT;

  int wave = threadIdx.x >> 5;
  int lane = threadIdx.x & 31;
  int m0 = blockIdx.x * 128 + (wave & 3) * 32;
  int n0 = blockIdx.y * 128 + (wave >> 2) * 64;

  v8f acc[2][4];
#pragma unroll
  for (int i = 0; i < 2; ++i)
#pragma unroll
    for (int j = 0; j < 4; ++j) acc[i][j] = (v8f)(0.f);

#pragma unroll
  for (int k0 = 0; k0 < DH; k0 += 32) {
    v16bf a0 = load_frag_b16(Qb, DH, m0,      k0);
    v16bf a1 = load_frag_b16(Qb, DH, m0 + 16, k0);
    v16bf b[4];
#pragma unroll
    for (int j = 0; j < 4; ++j) b[j] = load_frag_b16(Kb, DH, n0 + 16 * j, k0);
#pragma unroll
    for (int j = 0; j < 4; ++j) {
      acc[0][j] = WMMA_BF16(a0, b[j], acc[0][j]);
      acc[1][j] = WMMA_BF16(a1, b[j], acc[1][j]);
    }
  }

  const float sc = 0.125f;  // 1/sqrt(64)
  int nlo = lane & 15, rhi = (lane >> 4) * 8;
#pragma unroll
  for (int i = 0; i < 2; ++i)
#pragma unroll
    for (int j = 0; j < 4; ++j) {
      int n = n0 + 16 * j + nlo;
#pragma unroll
      for (int r = 0; r < 8; ++r) {
        int m = m0 + 16 * i + rhi + r;
        Sb[(size_t)m * TT + n] = acc[i][j][r] * sc;
      }
    }
}

// ---------------- row softmax over 2048 (in place on attn_weights) ----------------
__global__ __launch_bounds__(256) void softmax_kernel(float* __restrict__ attn) {
  float* p = attn + (size_t)blockIdx.x * TT;
  int tid = threadIdx.x;
  float v[8], mx = -3.4e38f;
#pragma unroll
  for (int i = 0; i < 8; ++i) { v[i] = p[tid + i * 256]; mx = fmaxf(mx, v[i]); }
  __shared__ float red[256];
  red[tid] = mx; __syncthreads();
  for (int s = 128; s > 0; s >>= 1) {
    if (tid < s) red[tid] = fmaxf(red[tid], red[tid + s]);
    __syncthreads();
  }
  mx = red[0]; __syncthreads();
  float sum = 0.f;
#pragma unroll
  for (int i = 0; i < 8; ++i) { v[i] = __expf(v[i] - mx); sum += v[i]; }
  red[tid] = sum; __syncthreads();
  for (int s = 128; s > 0; s >>= 1) {
    if (tid < s) red[tid] += red[tid + s];
    __syncthreads();
  }
  float inv = 1.0f / red[0];
#pragma unroll
  for (int i = 0; i < 8; ++i) p[tid + i * 256] = v[i] * inv;
}

// ---------------- PV: O = P @ V ----------------
// P (fp32, post-softmax) staged 128x32 per k-step into double-buffered LDS via
// async copies (coalesced 128B rows, loaded once per block instead of twice per
// wave-pair), V^T staged 64x32 bf16 alongside; fp32->bf16 conversion happens
// out of LDS right before the WMMAs.
__global__ __launch_bounds__(256) void pv_kernel(const float* __restrict__ attn,
                                                 const __bf16* __restrict__ Vt,
                                                 __bf16* __restrict__ AO) {
  __shared__ float  lP[2][128 * 32];   // 16 KiB per buffer
  __shared__ __bf16 lV[2][64 * 32];    // 4 KiB per buffer

  int bh = blockIdx.z;
  int b_ = bh >> 4, h = bh & (HH - 1);
  const float*  Pb = attn + (size_t)bh * TT * TT;
  const __bf16* Vb = Vt   + (size_t)bh * DH * TT;

  int tid  = threadIdx.x;
  int wave = tid >> 5;
  int lane = tid & 31;
  int mb = blockIdx.x * 128;
  int wm = (wave & 3) * 32;            // query rows within block tile
  int wn = (wave >> 2) * 32;           // head-dim cols (0 or 32)

  // Stage P (128x32 fp32 = 1024 chunks -> 4/thread) and V^T (64x32 bf16 =
  // 256 chunks -> 1/thread): 5 async copies per thread per k-step.
  auto prefetch = [&](int buf, int k0) {
#pragma unroll
    for (int i = 0; i < 4; ++i) {
      int c   = tid * 4 + i;
      int row = c >> 3;
      int col = (c & 7) * 4;
      const float* gp = Pb + (size_t)(mb + row) * TT + k0 + col;
      async_b128((unsigned)(size_t)&lP[buf][row * 32 + col], gp);
    }
    {
      int row = tid >> 2;
      int col = (tid & 3) * 8;
      const __bf16* gv = Vb + (size_t)row * TT + k0 + col;
      async_b128((unsigned)(size_t)&lV[buf][row * 32 + col], gv);
    }
  };

  v8f acc[2][2];
#pragma unroll
  for (int i = 0; i < 2; ++i)
#pragma unroll
    for (int j = 0; j < 2; ++j) acc[i][j] = (v8f)(0.f);

  const int KSTEPS = TT / 32;
  prefetch(0, 0);
  for (int kk = 0; kk < KSTEPS; ++kk) {
    int cur = kk & 1;
    if (kk + 1 < KSTEPS) {
      prefetch(cur ^ 1, (kk + 1) * 32);
      WAIT_ASYNC(0x5);                 // current batch (in-order) complete
    } else {
      WAIT_ASYNC(0x0);
    }
    __syncthreads();

    v16bf a0 = lds_frag_f32(lP[cur], wm);
    v16bf a1 = lds_frag_f32(lP[cur], wm + 16);
    v16bf b0 = lds_frag(lV[cur], wn);
    v16bf b1 = lds_frag(lV[cur], wn + 16);
    acc[0][0] = WMMA_BF16(a0, b0, acc[0][0]);
    acc[0][1] = WMMA_BF16(a0, b1, acc[0][1]);
    acc[1][0] = WMMA_BF16(a1, b0, acc[1][0]);
    acc[1][1] = WMMA_BF16(a1, b1, acc[1][1]);
    __syncthreads();
  }

  int m0 = mb + wm;
  int nlo = lane & 15, rhi = (lane >> 4) * 8;
#pragma unroll
  for (int i = 0; i < 2; ++i)
#pragma unroll
    for (int j = 0; j < 2; ++j) {
      int d = wn + 16 * j + nlo;
#pragma unroll
      for (int r = 0; r < 8; ++r) {
        int t = m0 + 16 * i + rhi + r;
        AO[((size_t)(b_ * TT + t)) * CC + h * DH + d] = f2bf(acc[i][j][r]);
      }
    }
}

// ---------------- host side ----------------
extern "C" void kernel_launch(void* const* d_in, const int* in_sizes, int n_in,
                              void* d_out, int out_size, void* d_ws, size_t ws_size,
                              hipStream_t stream) {
  const float* x     = (const float*)d_in[0];
  const float* gamma = (const float*)d_in[1];
  const float* beta  = (const float*)d_in[2];
  const float* wq = (const float*)d_in[3];  const float* bq = (const float*)d_in[4];
  const float* wk = (const float*)d_in[5];  const float* bk = (const float*)d_in[6];
  const float* wv = (const float*)d_in[7];  const float* bv = (const float*)d_in[8];
  const float* wo = (const float*)d_in[9];  const float* bo = (const float*)d_in[10];

  float* out  = (float*)d_out;               // [4096 x 1024]
  float* attn = (float*)d_out + ATTN_OFF;    // [32 x 2048 x 2048]

  // workspace layout (bytes)
  char* ws = (char*)d_ws;
  float*  scale = (float*)(ws + 0);
  float*  shift = (float*)(ws + 4096);
  __bf16* xn  = (__bf16*)(ws + 8192);                         // 8 MiB
  __bf16* wqb = (__bf16*)(ws + 8192 + 8388608);               // 2 MiB each
  __bf16* wkb = (__bf16*)(ws + 8192 + 8388608 + 2097152);
  __bf16* wvb = (__bf16*)(ws + 8192 + 8388608 + 2 * 2097152);
  __bf16* wob = (__bf16*)(ws + 8192 + 8388608 + 3 * 2097152);
  __bf16* Qb  = (__bf16*)(ws + 8192 + 8388608 + 4 * 2097152);            // 8 MiB
  __bf16* Kb  = (__bf16*)(ws + 8192 + 2 * 8388608 + 4 * 2097152);        // 8 MiB
  __bf16* Vt  = (__bf16*)(ws + 8192 + 3 * 8388608 + 4 * 2097152);        // 8 MiB
  __bf16* AO  = (__bf16*)(ws + 8192 + 4 * 8388608 + 4 * 2097152);        // 8 MiB

  // 1) BatchNorm stats + apply (emit bf16 normalized activations)
  bn_stats_kernel<<<CC, 256, 0, stream>>>(x, gamma, beta, scale, shift);
  bn_apply_kernel<<<(MTOK * CC) / 256, 256, 0, stream>>>(x, scale, shift, xn);

  // 2) weights -> bf16
  cvt_bf16_kernel<<<(CC * CC) / 256, 256, 0, stream>>>(wq, wqb, CC * CC);
  cvt_bf16_kernel<<<(CC * CC) / 256, 256, 0, stream>>>(wk, wkb, CC * CC);
  cvt_bf16_kernel<<<(CC * CC) / 256, 256, 0, stream>>>(wv, wvb, CC * CC);
  cvt_bf16_kernel<<<(CC * CC) / 256, 256, 0, stream>>>(wo, wob, CC * CC);

  // 3) Q/K/V projections (WMMA + async-LDS staging), V transposed
  dim3 gproj(MTOK / 128, CC / 128);
  gemm_proj_kernel<<<gproj, 256, 0, stream>>>(xn, wqb, bq, Qb, 0);
  gemm_proj_kernel<<<gproj, 256, 0, stream>>>(xn, wkb, bk, Kb, 1);
  gemm_proj_kernel<<<gproj, 256, 0, stream>>>(xn, wvb, bv, Vt, 2);

  // 4) scores (WMMA) -> attn_weights region, then row softmax in place
  dim3 gsc(TT / 128, TT / 128, BH);
  scores_kernel<<<gsc, 256, 0, stream>>>(Qb, Kb, attn);
  softmax_kernel<<<BH * TT, 256, 0, stream>>>(attn);

  // 5) attention output (WMMA + async-LDS staging): AO = P @ V
  dim3 gpv(TT / 128, 1, BH);
  pv_kernel<<<gpv, 256, 0, stream>>>(attn, Vt, AO);

  // 6) output projection (WMMA) -> fp32 out
  gemm_proj_kernel<<<gproj, 256, 0, stream>>>(AO, wob, bo, out, 3);
}
